// SSHEAggregatorLayer_74758200754738
// MI455X (gfx1250) — matrix operand order repaired and checked by reading, not compiled, not run on hardware
//
#include <hip/hip_runtime.h>
#include <hip/hip_bf16.h>
#include <stdint.h>

// out = 0.25 * ( fxp(xa)@fxp(wa) + fxp(xb)@fxp(wb) ) - 0.5 - y
// xa,xb (8192,2048) f32; wa,wb (2048,1024) f32; y,out (8192,1024) f32.
// Path A (ws large enough): pre-split to bf16 hi/lo planes in d_ws, then
//   WMMA GEMM with Tensor-Data-Mover (tensor_load_to_lds + s_wait_tensorcnt)
//   tile DMA, falling back to global_load_async_to_lds_b128 if the TDM
//   builtin is unavailable.
// Path B (fallback): fused conversion GEMM.

typedef __attribute__((ext_vector_type(16))) __bf16 v16bf;
typedef __attribute__((ext_vector_type(8)))  float  v8f;
typedef __attribute__((ext_vector_type(4)))  unsigned int v4u;
typedef __attribute__((ext_vector_type(8)))  int    v8i;
typedef __attribute__((ext_vector_type(4)))  int    v4i;

#define BM     128
#define BN     128
#define BK     32
#define PITCH  40          // bf16 per LDS row: 32 data + 8 pad (64B + 16B)
#define NROWS  8192
#define KHALF  2048
#define KFULL  4096
#define NCOLS  1024
#define KSTEPS 128         // 4096/32

union FragU { uint4 q[2]; v16bf v; };
union Pack8 { __bf16 b[8]; uint4 q; };

__device__ __forceinline__ float fxp(float x) {
  return rintf(x * 65536.0f) * (1.0f / 65536.0f);  // 2^-16 fixed-point grid
}
__device__ __forceinline__ void split_bf16(float x, __bf16& h, __bf16& l) {
  h = (__bf16)x;
  l = (__bf16)(x - (float)h);
}

// ---------------------------------------------------------------------------
// Pre-pass 1: X (row-major, K=2048 per source) -> Xhi/Xlo planes (8192 x 4096)
// ---------------------------------------------------------------------------
__global__ __launch_bounds__(256)
void split_x_kernel(const float* __restrict__ src, __bf16* __restrict__ hi,
                    __bf16* __restrict__ lo, int c0)
{
  const size_t g    = (size_t)blockIdx.x * 256 + threadIdx.x;
  const size_t base = g * 16;                       // 16 f32 per thread
  const size_t row  = base / KHALF;
  const int    col  = (int)(base % KHALF);
  const float4* sp  = (const float4*)(src + row * KHALF + col);
  float4 v[4];
  v[0] = sp[0]; v[1] = sp[1]; v[2] = sp[2]; v[3] = sp[3];
  const float* f = (const float*)v;
  const size_t dst = row * KFULL + c0 + col;
  #pragma unroll
  for (int j = 0; j < 2; ++j) {
    Pack8 ph, pl;
    #pragma unroll
    for (int e = 0; e < 8; ++e) {
      float xq = fxp(f[8 * j + e]);
      split_bf16(xq, ph.b[e], pl.b[e]);
    }
    *(uint4*)(hi + dst + 8 * j) = ph.q;
    *(uint4*)(lo + dst + 8 * j) = pl.q;
  }
}

// ---------------------------------------------------------------------------
// Pre-pass 2: W (2048 x 1024, row-major K x N) -> transposed Wt planes
//             Wt[n][k0 + k] (1024 x 4096), via 64x64 LDS transpose tile
// ---------------------------------------------------------------------------
__global__ __launch_bounds__(256)
void split_wt_kernel(const float* __restrict__ w, __bf16* __restrict__ hi,
                     __bf16* __restrict__ lo, int k0)
{
  __shared__ float tile[64 * 68];
  const int tid = threadIdx.x;
  const int kt  = blockIdx.x * 64;
  const int nb  = blockIdx.y * 64;
  {
    const int krow = tid >> 2;
    const int nch  = (tid & 3) * 16;
    const float4* sp = (const float4*)(w + (size_t)(kt + krow) * NCOLS + nb + nch);
    float* dp = &tile[krow * 68 + nch];
    #pragma unroll
    for (int i = 0; i < 4; ++i) ((float4*)dp)[i] = sp[i];
  }
  __syncthreads();
  {
    const int nrow = tid >> 2;
    const int kch  = (tid & 3) * 16;
    Pack8 ph[2], pl[2];
    #pragma unroll
    for (int i = 0; i < 16; ++i) {
      float xq = fxp(tile[(kch + i) * 68 + nrow]);
      split_bf16(xq, ph[i >> 3].b[i & 7], pl[i >> 3].b[i & 7]);
    }
    const size_t dst = (size_t)(nb + nrow) * KFULL + k0 + kt + kch;
    *(uint4*)(hi + dst)     = ph[0].q;
    *(uint4*)(hi + dst + 8) = ph[1].q;
    *(uint4*)(lo + dst)     = pl[0].q;
    *(uint4*)(lo + dst + 8) = pl[1].q;
  }
}

// ---------------------------------------------------------------------------
// CDNA5 tile DMA machinery
// ---------------------------------------------------------------------------
#if defined(__has_builtin)
#  if __has_builtin(__builtin_amdgcn_tensor_load_to_lds)
#    define HAVE_TDM 1
#  endif
#endif
#ifndef HAVE_TDM
#  define HAVE_TDM 0
#endif

#if HAVE_TDM
__device__ __forceinline__ void tdm_load(v4u g0, v8i g1) {
  v4i z4 = {0, 0, 0, 0};
#  if defined(__clang_major__) && (__clang_major__ >= 23)
  v8i z8 = {0, 0, 0, 0, 0, 0, 0, 0};
  __builtin_amdgcn_tensor_load_to_lds(g0, g1, z4, z4, z8, 0);
#  else
  __builtin_amdgcn_tensor_load_to_lds(g0, g1, z4, z4, 0);
#  endif
}
__device__ __forceinline__ void wait_tensor_ahead() {
#  if __has_builtin(__builtin_amdgcn_s_wait_tensorcnt)
  __builtin_amdgcn_s_wait_tensorcnt(1);
#  else
  asm volatile("s_wait_tensorcnt 1" ::: "memory");
#  endif
}
__device__ __forceinline__ void wait_tensor_zero() {
#  if __has_builtin(__builtin_amdgcn_s_wait_tensorcnt)
  __builtin_amdgcn_s_wait_tensorcnt(0);
#  else
  asm volatile("s_wait_tensorcnt 0" ::: "memory");
#  endif
}
#else
__device__ __forceinline__ void async_ld16(uint32_t lds_off, const void* gptr) {
  asm volatile("global_load_async_to_lds_b128 %0, %1, off"
               :: "v"(lds_off), "v"(gptr) : "memory");
}
#define WAIT_ASYNC_8() asm volatile("s_wait_asynccnt 8" ::: "memory")
#define WAIT_ASYNC_0() asm volatile("s_wait_asynccnt 0" ::: "memory")
#endif

// ---------------------------------------------------------------------------
// Main GEMM (two-pass path): bf16 hi/lo planes, DMA double-buffered LDS
// ---------------------------------------------------------------------------
#define PLANE      (BM * PITCH)                     // 5120 bf16 = 10240 B
#define PLANE_B    (PLANE * 2)                      // bytes
#define BUFSTRIDE  (4 * PLANE)                      // Ahi,Alo,Bhi,Blo (bf16)
#define SMEM_BYTES (2 * BUFSTRIDE * 2)              // 81920 B dynamic LDS

__global__ __launch_bounds__(256)
void gemm_async_kernel(const __bf16* __restrict__ Xhi, const __bf16* __restrict__ Xlo,
                       const __bf16* __restrict__ Wthi, const __bf16* __restrict__ Wtlo,
                       const float* __restrict__ y, float* __restrict__ out)
{
  extern __shared__ __align__(16) __bf16 smem[];    // [2][4][PLANE]

  const int tid  = threadIdx.x;
  const int lane = tid & 31;
  const int wave = tid >> 5;
  const int wm   = wave >> 1;                       // 0..3
  const int wn   = wave & 1;                        // 0..1
  const int m0   = blockIdx.y * BM;
  const int n0   = blockIdx.x * BN;

#if HAVE_TDM
  // Waves 0..3 each own one plane (Ahi,Alo,Bhi,Blo) and drive the TDM.
  // D#: 2D tile 32x128 elems (2B), row 64B + LDS pad 16B (pitch = 80B),
  //     tensor_dim0_stride = 4096 elems. All wave-uniform -> SGPR descriptor.
  const bool issuer = (wave < 4);
  const __bf16* tsrc =
      (wave & 2) ? ((wave & 1) ? Wtlo : Wthi) : ((wave & 1) ? Xlo : Xhi);
  const int trow = (wave & 2) ? n0 : m0;
  const uint32_t lds0 =
      (uint32_t)(uintptr_t)smem + (uint32_t)(wave & 3) * PLANE_B;

  auto issue = [&](int kk) {
    if (!issuer) return;
    const uint32_t lds = lds0 + (uint32_t)(kk & 1) * (BUFSTRIDE * 2);
    const uint64_t ga =
        (uint64_t)(uintptr_t)(tsrc + (size_t)trow * KFULL + (size_t)kk * BK);
    v4u g0 = { 1u,                                   // count=1, user desc
               lds,                                  // lds_addr (bytes)
               (uint32_t)ga,                         // global_addr[31:0]
               (uint32_t)((ga >> 32) & 0x01FFFFFFu) | 0x80000000u }; // type=2
    v8i g1 = { (int)0x06D10000u,   // data_size=2B, pad_en, interval=16dw, amount=4dw
               (int)0x10000000u,   // tensor_dim0 = 4096 (low16 in [31:16])
               (int)0x20000000u,   // tensor_dim1 = 8192 (low16 in [31:16])
               (int)(32u << 16),   // tile_dim0 = 32 elems
               (int)128,           // tile_dim1 = 128 rows
               (int)4096,          // tensor_dim0_stride = 4096 elems
               0, 0 };
    tdm_load(g0, g1);
  };
#else
  // Per-lane async copy: 512 x 16B chunks per plane pair; 2 chunks/thread
  auto issue = [&](int kk) {
    __bf16* buf = smem + (size_t)(kk & 1) * BUFSTRIDE;
    const int kb = kk * BK;
    #pragma unroll
    for (int h = 0; h < 2; ++h) {
      const int c   = tid + 256 * h;
      const int row = c >> 2;
      const int ko  = (c & 3) * 8;
      const uint32_t dA = (uint32_t)(uintptr_t)(buf + row * PITCH + ko);
      const uint32_t dB = (uint32_t)(uintptr_t)(buf + 2 * PLANE + row * PITCH + ko);
      async_ld16(dA,           Xhi  + (size_t)(m0 + row) * KFULL + kb + ko);
      async_ld16(dA + PLANE_B, Xlo  + (size_t)(m0 + row) * KFULL + kb + ko);
      async_ld16(dB,           Wthi + (size_t)(n0 + row) * KFULL + kb + ko);
      async_ld16(dB + PLANE_B, Wtlo + (size_t)(n0 + row) * KFULL + kb + ko);
    }
  };
#endif

  v8f acc[2][4];
  #pragma unroll
  for (int i = 0; i < 2; ++i)
    #pragma unroll
    for (int j = 0; j < 4; ++j)
      acc[i][j] = (v8f){0.f, 0.f, 0.f, 0.f, 0.f, 0.f, 0.f, 0.f};

  issue(0);
  issue(1);

  const int l15 = lane & 15;
  #pragma unroll 1
  for (int kk = 0; kk < KSTEPS; ++kk) {
#if HAVE_TDM
    if (issuer) {
      if (kk + 1 < KSTEPS) wait_tensor_ahead(); else wait_tensor_zero();
    }
#else
    if (kk + 1 < KSTEPS) WAIT_ASYNC_8(); else WAIT_ASYNC_0();
#endif
    __syncthreads();

    const __bf16* buf = smem + (size_t)(kk & 1) * BUFSTRIDE;
    const __bf16* sAhi = buf;
    const __bf16* sAlo = buf + PLANE;
    const __bf16* sBhi = buf + 2 * PLANE;
    const __bf16* sBlo = buf + 3 * PLANE;

    // A fragments: lanes 0-15 K{0..7,16..23}; lanes 16-31 K{8..15,24..31}
    FragU fah[2], fal[2];
    const int khalf = (lane >> 4) * 8;
    #pragma unroll
    for (int im = 0; im < 2; ++im) {
      const int r = (wm * 32 + im * 16 + l15) * PITCH;
      fah[im].q[0] = *(const uint4*)&sAhi[r + khalf];
      fah[im].q[1] = *(const uint4*)&sAhi[r + 16 + khalf];
      fal[im].q[0] = *(const uint4*)&sAlo[r + khalf];
      fal[im].q[1] = *(const uint4*)&sAlo[r + 16 + khalf];
    }
    // B fragments per n-tile: lanes 0-15 K{0..7,16..23}; lanes 16-31 K{8..15,24..31}
    const int kb2 = (lane >> 4) * 8;
    #pragma unroll
    for (int in = 0; in < 4; ++in) {
      FragU fbh, fbl;
      const int c = (wn * 64 + in * 16 + l15) * PITCH;
      fbh.q[0] = *(const uint4*)&sBhi[c + kb2];
      fbh.q[1] = *(const uint4*)&sBhi[c + kb2 + 16];
      fbl.q[0] = *(const uint4*)&sBlo[c + kb2];
      fbl.q[1] = *(const uint4*)&sBlo[c + kb2 + 16];
      #pragma unroll
      for (int im = 0; im < 2; ++im) {
        v8f cc = acc[im][in];
        cc = __builtin_amdgcn_wmma_f32_16x16x32_bf16(false, fah[im].v, false, fbh.v,
                                                     (short)0, cc, false, false);
        cc = __builtin_amdgcn_wmma_f32_16x16x32_bf16(false, fah[im].v, false, fbl.v,
                                                     (short)0, cc, false, false);
        cc = __builtin_amdgcn_wmma_f32_16x16x32_bf16(false, fal[im].v, false, fbh.v,
                                                     (short)0, cc, false, false);
        acc[im][in] = cc;
      }
    }
    __syncthreads();
    if (kk + 2 < KSTEPS) issue(kk + 2);
  }

  // epilogue: z = 0.25*s - 0.5 - y
  const int rsel = (lane >> 4) * 8;
  #pragma unroll
  for (int im = 0; im < 2; ++im) {
    const int rowbase = m0 + wm * 32 + im * 16 + rsel;
    #pragma unroll
    for (int in = 0; in < 4; ++in) {
      const int col = n0 + wn * 64 + in * 16 + l15;
      v8f c = acc[im][in];
      #pragma unroll
      for (int v = 0; v < 8; ++v) {
        const size_t idx = (size_t)(rowbase + v) * NCOLS + col;
        out[idx] = 0.25f * c[v] - 0.5f - y[idx];
      }
    }
  }
}

// ---------------------------------------------------------------------------
// Fallback: fused conversion GEMM (used only if d_ws is too small)
// ---------------------------------------------------------------------------
__global__ __launch_bounds__(256)
void gemm_fused_kernel(const float* __restrict__ xa, const float* __restrict__ xb,
                       const float* __restrict__ y,
                       const float* __restrict__ wa, const float* __restrict__ wb,
                       float* __restrict__ out)
{
  __shared__ alignas(16) __bf16 sAhi[BM * PITCH];
  __shared__ alignas(16) __bf16 sAlo[BM * PITCH];
  __shared__ alignas(16) __bf16 sBhi[BN * PITCH];
  __shared__ alignas(16) __bf16 sBlo[BN * PITCH];

  const int tid  = threadIdx.x;
  const int lane = tid & 31;
  const int wave = tid >> 5;
  const int wm   = wave >> 1;
  const int wn   = wave & 1;
  const int m0   = blockIdx.y * BM;
  const int n0   = blockIdx.x * BN;

  const int arow = tid >> 1;
  const int acol = (tid & 1) * 16;
  const int bk   = tid >> 3;
  const int bc0  = (tid & 7) * 16;

  v8f acc[2][4];
  for (int i = 0; i < 2; ++i)
    for (int j = 0; j < 4; ++j)
      acc[i][j] = (v8f){0.f, 0.f, 0.f, 0.f, 0.f, 0.f, 0.f, 0.f};

  const int l15 = lane & 15;
  #pragma unroll 1
  for (int kk = 0; kk < 2 * (KHALF / BK); ++kk) {
    const int phase = kk >> 6;
    const int kb    = (kk & 63) * BK;
    const float* x  = phase ? xb : xa;
    const float* w  = phase ? wb : wa;
    {
      const float4* ap = (const float4*)(x + (size_t)(m0 + arow) * KHALF + kb + acol);
      #pragma unroll
      for (int j = 0; j < 2; ++j) {
        float4 v0 = ap[2 * j], v1 = ap[2 * j + 1];
        const float* f = (const float*)&v0;
        Pack8 hi, lo;
        #pragma unroll
        for (int e = 0; e < 4; ++e) {
          float xq = fxp(f[e]); split_bf16(xq, hi.b[e], lo.b[e]);
        }
        const float* g = (const float*)&v1;
        #pragma unroll
        for (int e = 0; e < 4; ++e) {
          float xq = fxp(g[e]); split_bf16(xq, hi.b[4 + e], lo.b[4 + e]);
        }
        *(uint4*)&sAhi[arow * PITCH + acol + 8 * j] = hi.q;
        *(uint4*)&sAlo[arow * PITCH + acol + 8 * j] = lo.q;
      }
      const float4* bp = (const float4*)(w + (size_t)(kb + bk) * NCOLS + n0 + bc0);
      float4 bv[4];
      bv[0] = bp[0]; bv[1] = bp[1]; bv[2] = bp[2]; bv[3] = bp[3];
      const float* bf = (const float*)bv;
      #pragma unroll
      for (int e = 0; e < 16; ++e) {
        int ii = (e + ((tid & 7) << 1)) & 15;
        float wq = fxp(bf[ii]);
        __bf16 h, l; split_bf16(wq, h, l);
        sBhi[(bc0 + ii) * PITCH + bk] = h;
        sBlo[(bc0 + ii) * PITCH + bk] = l;
      }
    }
    __syncthreads();

    FragU fah[2], fal[2];
    const int khalf = (lane >> 4) * 8;
    #pragma unroll
    for (int im = 0; im < 2; ++im) {
      const int r = (wm * 32 + im * 16 + l15) * PITCH;
      fah[im].q[0] = *(const uint4*)&sAhi[r + khalf];
      fah[im].q[1] = *(const uint4*)&sAhi[r + 16 + khalf];
      fal[im].q[0] = *(const uint4*)&sAlo[r + khalf];
      fal[im].q[1] = *(const uint4*)&sAlo[r + 16 + khalf];
    }
    const int kb2 = (lane >> 4) * 8;
    #pragma unroll
    for (int in = 0; in < 4; ++in) {
      FragU fbh, fbl;
      const int c = (wn * 64 + in * 16 + l15) * PITCH;
      fbh.q[0] = *(const uint4*)&sBhi[c + kb2];
      fbh.q[1] = *(const uint4*)&sBhi[c + kb2 + 16];
      fbl.q[0] = *(const uint4*)&sBlo[c + kb2];
      fbl.q[1] = *(const uint4*)&sBlo[c + kb2 + 16];
      #pragma unroll
      for (int im = 0; im < 2; ++im) {
        v8f cc = acc[im][in];
        cc = __builtin_amdgcn_wmma_f32_16x16x32_bf16(false, fah[im].v, false, fbh.v,
                                                     (short)0, cc, false, false);
        cc = __builtin_amdgcn_wmma_f32_16x16x32_bf16(false, fah[im].v, false, fbl.v,
                                                     (short)0, cc, false, false);
        cc = __builtin_amdgcn_wmma_f32_16x16x32_bf16(false, fal[im].v, false, fbh.v,
                                                     (short)0, cc, false, false);
        acc[im][in] = cc;
      }
    }
    __syncthreads();
  }

  const int rsel = (lane >> 4) * 8;
  #pragma unroll
  for (int im = 0; im < 2; ++im) {
    const int rowbase = m0 + wm * 32 + im * 16 + rsel;
    #pragma unroll
    for (int in = 0; in < 4; ++in) {
      const int col = n0 + wn * 64 + in * 16 + l15;
      v8f c = acc[im][in];
      #pragma unroll
      for (int v = 0; v < 8; ++v) {
        const size_t idx = (size_t)(rowbase + v) * NCOLS + col;
        out[idx] = 0.25f * c[v] - 0.5f - y[idx];
      }
    }
  }
}

// ---------------------------------------------------------------------------
extern "C" void kernel_launch(void* const* d_in, const int* in_sizes, int n_in,
                              void* d_out, int out_size, void* d_ws, size_t ws_size,
                              hipStream_t stream) {
  (void)in_sizes; (void)n_in; (void)out_size;
  const float* xa = (const float*)d_in[0];
  const float* xb = (const float*)d_in[1];
  const float* y  = (const float*)d_in[2];
  const float* wa = (const float*)d_in[3];
  const float* wb = (const float*)d_in[4];
  float* out = (float*)d_out;

  const size_t nX  = (size_t)NROWS * KFULL;
  const size_t nWt = (size_t)NCOLS * KFULL;
  const size_t need = (2 * nX + 2 * nWt) * sizeof(__bf16);   // ~151 MB

  dim3 ggrid(NCOLS / BN, NROWS / BM);               // (8, 64)

  if (ws_size >= need) {
    __bf16* Xhi  = (__bf16*)d_ws;
    __bf16* Xlo  = Xhi + nX;
    __bf16* Wthi = Xlo + nX;
    __bf16* Wtlo = Wthi + nWt;

    const int xblocks = (int)(((size_t)NROWS * KHALF) / (256 * 16));  // 4096
    split_x_kernel<<<xblocks, 256, 0, stream>>>(xa, Xhi, Xlo, 0);
    split_x_kernel<<<xblocks, 256, 0, stream>>>(xb, Xhi, Xlo, KHALF);
    dim3 wgrid(KHALF / 64, NCOLS / 64);             // (32, 16)
    split_wt_kernel<<<wgrid, 256, 0, stream>>>(wa, Wthi, Wtlo, 0);
    split_wt_kernel<<<wgrid, 256, 0, stream>>>(wb, Wthi, Wtlo, KHALF);

    gemm_async_kernel<<<ggrid, 256, SMEM_BYTES, stream>>>(Xhi, Xlo, Wthi, Wtlo, y, out);
  } else {
    gemm_fused_kernel<<<ggrid, 256, 0, stream>>>(xa, xb, y, wa, wb, out);
  }
}
